// TreeLSTMCell_88210038325567
// MI455X (gfx1250) — matrix operand order repaired and checked by reading, not compiled, not run on hardware
//
#include <hip/hip_runtime.h>

typedef __bf16 v8bf  __attribute__((ext_vector_type(8)));
typedef __bf16 v16bf __attribute__((ext_vector_type(16)));
typedef float  v8f   __attribute__((ext_vector_type(8)));

#define KDIM 1536   // 256 (x part) + 1280 (h part)
#define CDIM 2048   // 1280 (f gates) + 768 (iou)

__device__ __forceinline__ unsigned short f2bf(float f) {
    unsigned u = __builtin_bit_cast(unsigned, f);
    u += 0x7FFFu + ((u >> 16) & 1u);          // round-to-nearest-even
    return (unsigned short)(u >> 16);
}

__device__ __forceinline__ v16bf cat8(v8bf lo, v8bf hi) {
    return __builtin_shufflevector(lo, hi, 0,1,2,3,4,5,6,7,8,9,10,11,12,13,14,15);
}

__device__ __forceinline__ float sigm(float v) { return 1.f / (1.f + __expf(-v)); }

// ---------------- pack A = [x*mask | neighbour_h] as bf16, zero-padded ----------------
__global__ void pack_a(const float* __restrict__ x, const float* __restrict__ mask,
                       const float* __restrict__ nh, unsigned short* __restrict__ A,
                       int N, int Npad) {
    size_t idx = (size_t)blockIdx.x * blockDim.x + threadIdx.x;
    size_t total = (size_t)Npad * KDIM;
    if (idx >= total) return;
    int row = (int)(idx / KDIM);
    int col = (int)(idx % KDIM);
    float v = 0.f;
    if (row < N) {
        if (col < 256) v = x[(size_t)row * 256 + col] * mask[row];
        else           v = nh[(size_t)row * 1280 + (col - 256)];
    }
    A[idx] = f2bf(v);
}

// ---------------- pack B^T [2048 cols][1536 k] bf16 ----------------
// cols 0..1279  : rows 0..255 = 2*W_f_in[k][c%256], rows 256..1535 = W_aggr_f[k-256][c]
// cols 1280..2047: rows 0..255 = W_iou_in[k][c-1280], rows 256..1535 = W_aggr_iou[k-256][c-1280]
__global__ void pack_b(const float* __restrict__ Wiou, const float* __restrict__ Wf,
                       const float* __restrict__ Waiou, const float* __restrict__ Waf,
                       unsigned short* __restrict__ BT) {
    size_t idx = (size_t)blockIdx.x * blockDim.x + threadIdx.x;
    if (idx >= (size_t)CDIM * KDIM) return;
    int c = (int)(idx / KDIM);
    int k = (int)(idx % KDIM);
    float v;
    if (c < 1280) {
        if (k < 256) v = 2.0f * Wf[(size_t)k * 256 + (c & 255)];
        else         v = Waf[(size_t)(k - 256) * 1280 + c];
    } else {
        int cc = c - 1280;
        if (k < 256) v = Wiou[(size_t)k * 768 + cc];
        else         v = Waiou[(size_t)(k - 256) * 768 + cc];
    }
    BT[idx] = f2bf(v);
}

// ---------------- fused GEMM + TreeLSTM epilogue ----------------
// Grid: Npad/32 workgroups, 512 threads (16 waves).
// wave w: mt = w&1 selects M-tile (16 rows), wx = w>>1 selects 256-col slab.
// wx 0..4 = forget-gate cols (child wx), wx 5/6/7 = i/o/u.
__launch_bounds__(512)
__global__ void treelstm_gemm(const unsigned short* __restrict__ Aus,
                              const unsigned short* __restrict__ BTus,
                              const float* __restrict__ mask,
                              const float* __restrict__ nC,
                              const float* __restrict__ b_iou_in,
                              const float* __restrict__ b_f_in,
                              const float* __restrict__ b_aggr_iou,
                              const float* __restrict__ b_aggr_f,
                              float* __restrict__ out, int N) {
    __shared__ float lds_ca[32 * 256];   // c_aggr accumulation
    __shared__ float lds_i [32 * 256];   // sigmoid(i)
    __shared__ float lds_o [32 * 256];   // sigmoid(o)

    const int tid = threadIdx.x;
    for (int q = tid; q < 32 * 256; q += 512) lds_ca[q] = 0.f;

    const int lane = tid & 31;
    const int w    = tid >> 5;
    const int mt   = w & 1;
    const int wx   = w >> 1;
    const int half = lane >> 4;
    const int ncl  = lane & 15;
    const int blockM = blockIdx.x * 32;
    const int m0w    = blockM + mt * 16;

    const __bf16* A  = reinterpret_cast<const __bf16*>(Aus);
    const __bf16* BT = reinterpret_cast<const __bf16*>(BTus);

    // A frag: lane = M row (both halves); half0 holds K{0..7,16..23}, half1 K{8..15,24..31}
    const __bf16* Abase = A  + (size_t)(m0w + ncl) * KDIM + half * 8;
    // B frag: lane = N col (both halves); half0 holds K{0..7,16..23}, half1 K{8..15,24..31}
    const __bf16* Bbase = BT + (size_t)(wx * 256 + ncl) * KDIM + half * 8;

    v8f acc[16];
#pragma unroll
    for (int t = 0; t < 16; ++t) acc[t] = v8f{0.f,0.f,0.f,0.f,0.f,0.f,0.f,0.f};

    for (int k0 = 0; k0 < KDIM; k0 += 32) {
        v8bf alo = *(const v8bf*)(Abase + k0);
        v8bf ahi = *(const v8bf*)(Abase + k0 + 16);
        v16bf af = cat8(alo, ahi);
        const __bf16* bp = Bbase + k0;
#pragma unroll
        for (int t = 0; t < 16; ++t) {
            v8bf blo = *(const v8bf*)(bp);
            v8bf bhi = *(const v8bf*)(bp + 16);
            v16bf bfr = cat8(blo, bhi);
            acc[t] = __builtin_amdgcn_wmma_f32_16x16x32_bf16(
                false, af, false, bfr, (short)0, acc[t], false, false);
            bp += (size_t)16 * KDIM;
        }
    }

    __syncthreads();  // orders lds_ca zero-init vs. epilogue too

    // accumulator layout: element r, lane L -> row = r + 8*(L>>4), col = L&15 (within tile)
    const int rowbase = mt * 16 + 8 * half;

    if (wx < 5) {                         // forget gates for child wx
        for (int r = 0; r < 8; ++r) {
            int n  = blockM + rowbase + r;
            bool ok = n < N;
            float mk = ok ? mask[n] : 0.f;
            int rw = rowbase + r;
#pragma unroll
            for (int t = 0; t < 16; ++t) {
                int j = t * 16 + ncl;
                int c = wx * 256 + j;
                float v  = acc[t][r] + b_aggr_f[c] + 2.f * mk * b_f_in[j];
                float fg = sigm(v);
                if (ok) {
                    float ctr = fg * nC[(size_t)n * 1280 + c];
                    atomicAdd(&lds_ca[rw * 256 + j], ctr);   // ds_add_f32
                }
            }
        }
    } else if (wx == 5 || wx == 6) {      // i / o gates
        int base = (wx - 5) * 256;
        float* ldst = (wx == 5) ? lds_i : lds_o;
        for (int r = 0; r < 8; ++r) {
            int n  = blockM + rowbase + r;
            bool ok = n < N;
            float mk = ok ? mask[n] : 0.f;
            int rw = rowbase + r;
#pragma unroll
            for (int t = 0; t < 16; ++t) {
                int j  = t * 16 + ncl;
                int cc = base + j;
                float v = acc[t][r] + b_aggr_iou[cc] + mk * b_iou_in[cc];
                if (ok) ldst[rw * 256 + j] = sigm(v);
            }
        }
    }
    __syncthreads();

    if (wx == 7) {                        // u gate + final combine
        size_t NH = (size_t)N * 256;
        for (int r = 0; r < 8; ++r) {
            int n = blockM + rowbase + r;
            if (n >= N) continue;
            float mk = mask[n];
            int rw = rowbase + r;
#pragma unroll
            for (int t = 0; t < 16; ++t) {
                int j  = t * 16 + ncl;
                int cc = 512 + j;
                float v = acc[t][r] + b_aggr_iou[cc] + mk * b_iou_in[cc];
                float u = tanhf(v);
                float cv = lds_i[rw * 256 + j] * u + lds_ca[rw * 256 + j];
                float h  = lds_o[rw * 256 + j] * tanhf(cv);
                out[(size_t)n * 256 + j] = h;
                out[NH + (size_t)n * 256 + j] = cv;
            }
        }
    }
}

extern "C" void kernel_launch(void* const* d_in, const int* in_sizes, int n_in,
                              void* d_out, int out_size, void* d_ws, size_t ws_size,
                              hipStream_t stream) {
    const float* x     = (const float*)d_in[0];
    const float* xmask = (const float*)d_in[1];
    const float* nh    = (const float*)d_in[2];
    const float* nc    = (const float*)d_in[3];
    const float* Wiou  = (const float*)d_in[4];
    const float* biou  = (const float*)d_in[5];
    const float* Wf    = (const float*)d_in[6];
    const float* bf    = (const float*)d_in[7];
    const float* Waiou = (const float*)d_in[8];
    const float* baiou = (const float*)d_in[9];
    const float* Waf   = (const float*)d_in[10];
    const float* baf   = (const float*)d_in[11];

    const int N    = in_sizes[1];           // x_mask has N elements
    const int Npad = ((N + 31) / 32) * 32;

    unsigned short* Aws  = (unsigned short*)d_ws;                  // Npad*1536 bf16
    unsigned short* BTws = Aws + (size_t)Npad * KDIM;              // 2048*1536 bf16

    size_t totalA = (size_t)Npad * KDIM;
    pack_a<<<(unsigned)((totalA + 255) / 256), 256, 0, stream>>>(x, xmask, nh, Aws, N, Npad);

    size_t totalB = (size_t)CDIM * KDIM;
    pack_b<<<(unsigned)((totalB + 255) / 256), 256, 0, stream>>>(Wiou, Wf, Waiou, Waf, BTws);

    treelstm_gemm<<<Npad / 32, 512, 0, stream>>>(
        Aws, BTws, xmask, nc, biou, bf, baiou, baf, (float*)d_out, N);
}
